// MultiheadAttention_15006615732527
// MI455X (gfx1250) — matrix operand order repaired and checked by reading, not compiled, not run on hardware
//
#include <hip/hip_runtime.h>
#include <stdint.h>

// ---------------- types ----------------
typedef __attribute__((ext_vector_type(16))) __bf16 bf16x16;
typedef __attribute__((ext_vector_type(8)))  __bf16 bf16x8;
typedef __attribute__((ext_vector_type(8)))  float  floatx8;
typedef __attribute__((ext_vector_type(4)))  unsigned int v4u;
typedef __attribute__((ext_vector_type(4)))  int v4i;
typedef __attribute__((ext_vector_type(8)))  int v8i;

#define D_MODEL 1024
#define NQKV    3072
#define WIN     128
#define DISP    64
#define SEQ     8192
#define BATCH   4
#define ROWS    (BATCH*SEQ)   // 32768
#define NWIN    (ROWS/WIN)    // 256
#define HEADS   16
#define DK      64

#if __has_builtin(__builtin_amdgcn_tensor_load_to_lds)
#define USE_TDM 1
#else
#define USE_TDM 0
#endif
#if __has_include(<hip/amd_detail/amd_gfx1250_TDM.h>)
#define TDM_6ARG 1
#endif

// float -> bf16 bits, round-to-nearest-even
static __device__ __forceinline__ unsigned short f2bf(float f) {
  union { float f; unsigned u; } a; a.f = f;
  unsigned u = a.u;
  u = u + 0x7FFFu + ((u >> 16) & 1u);
  return (unsigned short)(u >> 16);
}

// A-matrix 16x32 bf16 fragment: 16 contiguous K values per lane.
static __device__ __forceinline__ bf16x16 frag_a(const unsigned short* p) {
  return *(const bf16x16*)(p);
}
// B-matrix 32x16 bf16 fragment: 16 contiguous K values per lane.
static __device__ __forceinline__ bf16x16 frag_b(const unsigned short* p) {
  return *(const bf16x16*)(p);
}

#if USE_TDM
// Issue a TDM 2-D tile load: tile_d0 elems (contig, bf16) x tile_d1 rows,
// global row stride row_stride elems, into LDS at lds_off (contiguous).
// D# packing per CDNA5 ISA ch.8 (group0: count/lds/global/type; group1:
// data_size, tensor dims, tile dims, dim0 stride; groups 2/3 zero => 2-D).
static __device__ __forceinline__ void tdm_load_2d(unsigned lds_off,
                                                   const void* gptr,
                                                   unsigned tile_d0,
                                                   unsigned tile_d1,
                                                   unsigned long long row_stride) {
  const unsigned long long ga = (unsigned long long)(uintptr_t)gptr;
  v4u g0;
  g0[0] = 1u;                                   // count=1, user descriptor
  g0[1] = lds_off;                              // LDS byte address
  g0[2] = (unsigned)(ga & 0xFFFFFFFFu);         // global_addr[31:0]
  g0[3] = (unsigned)((ga >> 32) & 0x1FFFFFFu)   // global_addr[56:32]
        | (2u << 30);                           // type=2 ("image")
  const unsigned td0 = 0x40000000u;             // tensor_dim0 (no OOB clamp)
  const unsigned td1 = 0x40000000u;             // tensor_dim1
  v8i g1;
  g1[0] = (int)(1u << 16);                      // wg_mask=0, data_size=1 (2B)
  g1[1] = (int)((td0 & 0xFFFFu) << 16);         // [63:48]=tensor_dim0 lo
  g1[2] = (int)((td0 >> 16) | ((td1 & 0xFFFFu) << 16)); // dim0 hi | dim1 lo
  g1[3] = (int)((td1 >> 16) | (tile_d0 << 16)); // dim1 hi | tile_dim0
  g1[4] = (int)(tile_d1 & 0xFFFFu);             // tile_dim1 | tile_dim2=0
  g1[5] = (int)(row_stride & 0xFFFFFFFFu);      // tensor_dim0_stride[31:0]
  g1[6] = (int)((row_stride >> 32) & 0xFFFFu);  // stride[47:32] | dim1str lo=0
  g1[7] = 0;
  v4i z4 = (v4i)0;
#if defined(TDM_6ARG)
  v8i z8 = (v8i)0;
  __builtin_amdgcn_tensor_load_to_lds(g0, g1, z4, z4, z8, 0);
#else
  __builtin_amdgcn_tensor_load_to_lds(g0, g1, z4, z4, 0);
#endif
}
#endif

// ---------------- shared GEMM mainloop: C(128x128) += A(128x1024)*B^T ------
// aG/bG: row-0 pointers, row stride D_MODEL elems. sAb/sBb: [2][128*64] LDS.
static __device__ __forceinline__ void gemm128_mainloop(
    const unsigned short* __restrict__ aG, const unsigned short* __restrict__ bG,
    unsigned short* sAb, unsigned short* sBb,
    int tid, int wave, int lane, floatx8 (&acc)[2][4]) {
  const int wm = wave >> 1, wn = wave & 1;
  const int hi8  = (lane >> 4) << 3;
  const int hi16 = (lane >> 4) << 4;

#if USE_TDM
  if (wave == 0) {   // wave-uniform: one TDM issue per descriptor
    tdm_load_2d((unsigned)(uintptr_t)sAb, aG, 64, 128, D_MODEL);
    tdm_load_2d((unsigned)(uintptr_t)sBb, bG, 64, 128, D_MODEL);
  }
  for (int it = 0; it < 16; ++it) {
    const int cur = it & 1;
    unsigned short* sA = sAb + cur * (128 * 64);
    unsigned short* sB = sBb + cur * (128 * 64);
    if (wave == 0) {
      if (it + 1 < 16) {     // prefetch next K-slab into the other buffer
        const int k0n = (it + 1) * 64;
        tdm_load_2d((unsigned)(uintptr_t)(sAb + (cur ^ 1) * (128 * 64)),
                    aG + k0n, 64, 128, D_MODEL);
        tdm_load_2d((unsigned)(uintptr_t)(sBb + (cur ^ 1) * (128 * 64)),
                    bG + k0n, 64, 128, D_MODEL);
        __builtin_amdgcn_s_wait_tensorcnt(2);  // current buffer complete
      } else {
        __builtin_amdgcn_s_wait_tensorcnt(0);
      }
    }
    __syncthreads();
#pragma unroll
    for (int kt = 0; kt < 2; ++kt) {
      bf16x16 af[2];
#pragma unroll
      for (int mt = 0; mt < 2; ++mt)
        af[mt] = frag_a(sA + (wm * 32 + mt * 16 + (lane & 15)) * 64 + kt * 32 + hi16);
#pragma unroll
      for (int nt = 0; nt < 4; ++nt) {
        bf16x16 bf = frag_b(sB + (wn * 64 + nt * 16 + (lane & 15)) * 64 + kt * 32 + hi16);
#pragma unroll
        for (int mt = 0; mt < 2; ++mt)
          acc[mt][nt] = __builtin_amdgcn_wmma_f32_16x16x32_bf16(
              false, af[mt], false, bf, (short)0, acc[mt][nt], false, false);
      }
    }
    __syncthreads();   // protect buffer about to be overwritten by next TDM
  }
#else
  // Fallback: manual staging, loads batched before stores for MLP.
  const int r = tid >> 1, half = tid & 1;
  const unsigned short* aSrc = aG + (size_t)r * D_MODEL + half * 32;
  const unsigned short* bSrc = bG + (size_t)r * D_MODEL + half * 32;
  unsigned short* aDst = sAb + r * 64 + half * 32;
  unsigned short* bDst = sBb + r * 64 + half * 32;
  for (int k0 = 0; k0 < D_MODEL; k0 += 64) {
    uint4 ra[4], rb[4];
#pragma unroll
    for (int i = 0; i < 4; ++i) { ra[i] = ((const uint4*)(aSrc + k0))[i]; }
#pragma unroll
    for (int i = 0; i < 4; ++i) { rb[i] = ((const uint4*)(bSrc + k0))[i]; }
#pragma unroll
    for (int i = 0; i < 4; ++i) { ((uint4*)aDst)[i] = ra[i]; ((uint4*)bDst)[i] = rb[i]; }
    __syncthreads();
#pragma unroll
    for (int kt = 0; kt < 2; ++kt) {
      bf16x16 af[2];
#pragma unroll
      for (int mt = 0; mt < 2; ++mt)
        af[mt] = frag_a(sAb + (wm * 32 + mt * 16 + (lane & 15)) * 64 + kt * 32 + hi16);
#pragma unroll
      for (int nt = 0; nt < 4; ++nt) {
        bf16x16 bf = frag_b(sBb + (wn * 64 + nt * 16 + (lane & 15)) * 64 + kt * 32 + hi16);
#pragma unroll
        for (int mt = 0; mt < 2; ++mt)
          acc[mt][nt] = __builtin_amdgcn_wmma_f32_16x16x32_bf16(
              false, af[mt], false, bf, (short)0, acc[mt][nt], false, false);
      }
    }
    __syncthreads();
  }
#endif
}

// ---------------- kernel 1: roll(-DISP) + f32->bf16 ----------------
__global__ __launch_bounds__(256) void k_roll_bf16(const float* __restrict__ x,
                                                   unsigned short* __restrict__ xb) {
  const long long e = ((long long)blockIdx.x * 256 + threadIdx.x) * 8;
  const int r   = (int)(e >> 10);
  const int col = (int)(e & 1023);
  const int bi = r >> 13, si = r & (SEQ - 1);
  const int ss = (si + DISP) & (SEQ - 1);
  const float* src = x + ((long long)((bi << 13) + ss) << 10) + col;
  float4 f0 = ((const float4*)src)[0];
  float4 f1 = ((const float4*)src)[1];
  uint4 pk;
  pk.x = (unsigned)f2bf(f0.x) | ((unsigned)f2bf(f0.y) << 16);
  pk.y = (unsigned)f2bf(f0.z) | ((unsigned)f2bf(f0.w) << 16);
  pk.z = (unsigned)f2bf(f1.x) | ((unsigned)f2bf(f1.y) << 16);
  pk.w = (unsigned)f2bf(f1.z) | ((unsigned)f2bf(f1.w) << 16);
  *(uint4*)(xb + e) = pk;
}

// ---------------- kernel 2: transpose weights to n-major bf16 ----------------
__global__ __launch_bounds__(256) void k_wtrans(const float* __restrict__ wqkv,
                                                const float* __restrict__ wo,
                                                unsigned short* __restrict__ wqkvT,
                                                unsigned short* __restrict__ woT) {
  const int idx = blockIdx.x * 256 + threadIdx.x;
  if (idx < D_MODEL * NQKV) {
    const int k = idx / NQKV, n = idx % NQKV;
    wqkvT[(size_t)n * D_MODEL + k] = f2bf(wqkv[idx]);
  } else {
    const int j = idx - D_MODEL * NQKV;
    const int k = j >> 10, n = j & 1023;
    woT[(size_t)n * D_MODEL + k] = f2bf(wo[j]);
  }
}

// ---------------- kernel 3: QKV GEMM (128x128 tile, bf16 WMMA + TDM) --------
__global__ __launch_bounds__(256) void k_qkv_gemm(const unsigned short* __restrict__ xb,
    const unsigned short* __restrict__ wT, const float* __restrict__ bias,
    unsigned short* __restrict__ qd, unsigned short* __restrict__ kd,
    unsigned short* __restrict__ vd) {
  const int nb  = blockIdx.x;          // 0..23
  const int win = blockIdx.y;          // 0..255
  const int tid = threadIdx.x;
  const int lane = tid & 31, wave = tid >> 5;
  const int wm = wave >> 1, wn = wave & 1;
  const int hi8 = (lane >> 4) << 3;

  __shared__ alignas(32) unsigned short sA[2 * 128 * 64];
  __shared__ alignas(32) unsigned short sB[2 * 128 * 64];

  floatx8 acc[2][4];
#pragma unroll
  for (int a = 0; a < 2; ++a)
#pragma unroll
    for (int b = 0; b < 4; ++b) acc[a][b] = (floatx8)0.0f;

  gemm128_mainloop(xb + (size_t)(win * 128) * D_MODEL,
                   wT + (size_t)(nb * 128) * D_MODEL,
                   sA, sB, tid, wave, lane, acc);

  // epilogue: decode (head, which, f) column permutation; v stored transposed
#pragma unroll
  for (int nt = 0; nt < 4; ++nt) {
    const int c = nb * 128 + wn * 64 + nt * 16 + (lane & 15);
    const float bv = bias[c];
    const unsigned head  = (unsigned)c / 192u;
    const unsigned rr    = (unsigned)c - head * 192u;
    const unsigned which = rr >> 6;
    const unsigned f     = rr & 63u;
#pragma unroll
    for (int mt = 0; mt < 2; ++mt) {
#pragma unroll
      for (int i = 0; i < 8; ++i) {
        const int m = wm * 32 + mt * 16 + i + hi8;
        const unsigned short v16 = f2bf(acc[mt][nt][i] + bv);
        if (which == 0)
          qd[(((size_t)win * 16 + head) * 128 + m) * 64 + f] = v16;
        else if (which == 1)
          kd[(((size_t)win * 16 + head) * 128 + m) * 64 + f] = v16;
        else
          vd[(((size_t)win * 16 + head) * 64 + f) * 128 + m] = v16;  // v^T
      }
    }
  }
}

// ---------------- kernel 4: windowed attention, one block per (win, head) ----
__global__ __launch_bounds__(256) void k_attn(const unsigned short* __restrict__ qd,
    const unsigned short* __restrict__ kd, const unsigned short* __restrict__ vd,
    unsigned short* __restrict__ vals) {
  const int win  = blockIdx.x >> 4;
  const int head = blockIdx.x & 15;
  const int tid = threadIdx.x, lane = tid & 31, wave = tid >> 5;
  const int hi8  = (lane >> 4) << 3;
  const int hi16 = (lane >> 4) << 4;
  const int ncol0 = lane & 15;

  __shared__ alignas(32) unsigned short smem[128 * 64 + 128 * 64 + 64 * 128];
  unsigned short* qs  = smem;                 // [128][64]
  unsigned short* ks  = smem + 128 * 64;      // [128][64]
  unsigned short* vts = smem + 2 * 128 * 64;  // [64][128] = v^T
  unsigned short* ps  = smem;                 // [128][128], overlays qs+ks

  const size_t base = ((size_t)win * 16 + head) * 8192;
#if USE_TDM
  if (wave == 0) {   // three contiguous 16 KB DMAs
    tdm_load_2d((unsigned)(uintptr_t)qs,  qd + base, 8192, 1, 8192);
    tdm_load_2d((unsigned)(uintptr_t)ks,  kd + base, 8192, 1, 8192);
    tdm_load_2d((unsigned)(uintptr_t)vts, vd + base, 8192, 1, 8192);
    __builtin_amdgcn_s_wait_tensorcnt(0);
  }
#else
  {
    const uint4* qsrc = (const uint4*)(qd + base);
    const uint4* ksrc = (const uint4*)(kd + base);
    const uint4* vsrc = (const uint4*)(vd + base);
    uint4 rq[4], rk[4], rv[4];
#pragma unroll
    for (int i = 0; i < 4; ++i) {
      rq[i] = qsrc[tid + i * 256];
      rk[i] = ksrc[tid + i * 256];
      rv[i] = vsrc[tid + i * 256];
    }
#pragma unroll
    for (int i = 0; i < 4; ++i) {
      ((uint4*)qs)[tid + i * 256]  = rq[i];
      ((uint4*)ks)[tid + i * 256]  = rk[i];
      ((uint4*)vts)[tid + i * 256] = rv[i];
    }
  }
#endif
  __syncthreads();

  // scores: wave owns rows [wave*16, +16), all 128 columns
  floatx8 sc[8];
#pragma unroll
  for (int nt = 0; nt < 8; ++nt) sc[nt] = (floatx8)0.0f;
#pragma unroll
  for (int kt = 0; kt < 2; ++kt) {
    bf16x16 aq = frag_a(qs + (wave * 16 + (lane & 15)) * 64 + kt * 32 + hi16);
#pragma unroll
    for (int nt = 0; nt < 8; ++nt) {
      bf16x16 bk = frag_b(ks + (nt * 16 + (lane & 15)) * 64 + kt * 32 + hi16);
      sc[nt] = __builtin_amdgcn_wmma_f32_16x16x32_bf16(
          false, aq, false, bk, (short)0, sc[nt], false, false);
    }
  }

  // scale + shifted-window mask
#pragma unroll
  for (int nt = 0; nt < 8; ++nt) {
    const int n = nt * 16 + ncol0;
#pragma unroll
    for (int i = 0; i < 8; ++i) {
      const int m = wave * 16 + i + hi8;
      float v = sc[nt][i] * 0.125f;        // 1/sqrt(64)
      if ((m < 64) != (n < 64)) v = -9.0e15f;
      sc[nt][i] = v;
    }
  }

  // softmax per row: reduce over 8 tiles then across the 16-lane half-wave
  float inv[8];
#pragma unroll
  for (int i = 0; i < 8; ++i) {
    float mx = sc[0][i];
#pragma unroll
    for (int nt = 1; nt < 8; ++nt) mx = fmaxf(mx, sc[nt][i]);
#pragma unroll
    for (int d = 1; d < 16; d <<= 1) mx = fmaxf(mx, __shfl_xor(mx, d, 32));
    float sum = 0.f;
#pragma unroll
    for (int nt = 0; nt < 8; ++nt) {
      float e = __expf(sc[nt][i] - mx);
      sc[nt][i] = e;
      sum += e;
    }
#pragma unroll
    for (int d = 1; d < 16; d <<= 1) sum += __shfl_xor(sum, d, 32);
    inv[i] = 1.0f / sum;
  }
  __syncthreads();  // all waves done reading qs/ks before P overwrites them

#pragma unroll
  for (int nt = 0; nt < 8; ++nt) {
#pragma unroll
    for (int i = 0; i < 8; ++i) {
      const int m = wave * 16 + i + hi8;
      ps[m * 128 + nt * 16 + ncol0] = f2bf(sc[nt][i] * inv[i]);
    }
  }
  __syncthreads();

  // vals(16x64) = P(16x128) x V(128x64)
  floatx8 oacc[4];
#pragma unroll
  for (int nt = 0; nt < 4; ++nt) oacc[nt] = (floatx8)0.0f;
#pragma unroll
  for (int kk = 0; kk < 4; ++kk) {
    bf16x16 ap = frag_a(ps + (wave * 16 + (lane & 15)) * 128 + kk * 32 + hi16);
#pragma unroll
    for (int nt = 0; nt < 4; ++nt) {
      bf16x16 bv = frag_b(vts + (nt * 16 + (lane & 15)) * 128 + kk * 32 + hi16);
      oacc[nt] = __builtin_amdgcn_wmma_f32_16x16x32_bf16(
          false, ap, false, bv, (short)0, oacc[nt], false, false);
    }
  }
#pragma unroll
  for (int nt = 0; nt < 4; ++nt) {
    const int f = nt * 16 + ncol0;
#pragma unroll
    for (int i = 0; i < 8; ++i) {
      const int m = wave * 16 + i + hi8;
      vals[((size_t)win * 128 + m) * D_MODEL + head * 64 + f] = f2bf(oacc[nt][i]);
    }
  }
}

// ---------------- kernel 5: output GEMM + bias + roll(+DISP) ----------------
__global__ __launch_bounds__(256) void k_out_gemm(const unsigned short* __restrict__ vals,
    const unsigned short* __restrict__ woT, const float* __restrict__ bo,
    float* __restrict__ out) {
  const int nb = blockIdx.x;   // 0..7
  const int mb = blockIdx.y;   // 0..255
  const int tid = threadIdx.x;
  const int lane = tid & 31, wave = tid >> 5;
  const int wm = wave >> 1, wn = wave & 1;
  const int hi8 = (lane >> 4) << 3;

  __shared__ alignas(32) unsigned short sA[2 * 128 * 64];
  __shared__ alignas(32) unsigned short sB[2 * 128 * 64];

  floatx8 acc[2][4];
#pragma unroll
  for (int a = 0; a < 2; ++a)
#pragma unroll
    for (int b = 0; b < 4; ++b) acc[a][b] = (floatx8)0.0f;

  gemm128_mainloop(vals + (size_t)(mb * 128) * D_MODEL,
                   woT + (size_t)(nb * 128) * D_MODEL,
                   sA, sB, tid, wave, lane, acc);

#pragma unroll
  for (int nt = 0; nt < 4; ++nt) {
    const int c = nb * 128 + wn * 64 + nt * 16 + (lane & 15);
    const float bv = bo[c];
#pragma unroll
    for (int mt = 0; mt < 2; ++mt) {
#pragma unroll
      for (int i = 0; i < 8; ++i) {
        const int rg = mb * 128 + wm * 32 + mt * 16 + i + hi8;
        const int bi = rg >> 13;
        const int si = rg & (SEQ - 1);
        const int so = (si + DISP) & (SEQ - 1);   // roll(+DISP)
        out[((size_t)((bi << 13) + so)) * D_MODEL + c] = acc[mt][nt][i] + bv;
      }
    }
  }
}

// ---------------- host ----------------
extern "C" void kernel_launch(void* const* d_in, const int* in_sizes, int n_in,
                              void* d_out, int out_size, void* d_ws, size_t ws_size,
                              hipStream_t stream) {
  const float* x    = (const float*)d_in[0];
  const float* wqkv = (const float*)d_in[1];
  const float* bqkv = (const float*)d_in[2];
  const float* wo   = (const float*)d_in[3];
  const float* bo   = (const float*)d_in[4];
  float* out = (float*)d_out;

  char* ws = (char*)d_ws;
  size_t off = 0;
  auto wsalloc = [&](size_t bytes) -> char* {
    char* p = ws + off;
    off += (bytes + 255) & ~(size_t)255;
    return p;
  };
  unsigned short* xb    = (unsigned short*)wsalloc((size_t)ROWS * D_MODEL * 2);     // 64 MB
  unsigned short* wqkvT = (unsigned short*)wsalloc((size_t)NQKV * D_MODEL * 2);     //  6 MB
  unsigned short* woT   = (unsigned short*)wsalloc((size_t)D_MODEL * D_MODEL * 2);  //  2 MB
  unsigned short* qd    = (unsigned short*)wsalloc((size_t)NWIN * HEADS * WIN * DK * 2); // 64 MB
  unsigned short* kd    = (unsigned short*)wsalloc((size_t)NWIN * HEADS * WIN * DK * 2); // 64 MB
  unsigned short* vd    = (unsigned short*)wsalloc((size_t)NWIN * HEADS * WIN * DK * 2); // 64 MB
  unsigned short* vals  = (unsigned short*)wsalloc((size_t)ROWS * D_MODEL * 2);     // 64 MB

  k_roll_bf16<<<(ROWS * D_MODEL) / (8 * 256), 256, 0, stream>>>(x, xb);
  k_wtrans<<<(D_MODEL * NQKV + D_MODEL * D_MODEL) / 256, 256, 0, stream>>>(wqkv, wo, wqkvT, woT);
  k_qkv_gemm<<<dim3(NQKV / 128, NWIN), 256, 0, stream>>>(xb, wqkvT, bqkv, qd, kd, vd);
  k_attn<<<NWIN * HEADS, 256, 0, stream>>>(qd, kd, vd, vals);
  k_out_gemm<<<dim3(D_MODEL / 128, NWIN), 256, 0, stream>>>(vals, woT, bo, out);
}
